// WMSA_49830210568593
// MI455X (gfx1250) — matrix operand order repaired and checked by reading, compiled, not run on hardware
//
#include <hip/hip_runtime.h>

typedef __attribute__((ext_vector_type(16))) _Float16 v16h;
typedef __attribute__((ext_vector_type(8)))  _Float16 v8h;
typedef __attribute__((ext_vector_type(8)))  float    v8f;

#define CDIM   192
#define NQKV   576
#define NH     6
#define HD     32
#define PWIN   64
#define NWIN   1024      // 32x32 windows per image
#define SCALE_ 0.1767766952966369f   // 32^-0.5

// ---- WMMA helpers (CDNA5 gfx1250, wave32) -------------------------------
static __device__ __forceinline__ v8f wmma16(v16h a, v16h b, v8f c) {
  // D = A(16x32 f16) * B(32x16 f16) + C(16x16 f32)
  return __builtin_amdgcn_wmma_f32_16x16x32_f16(false, a, false, b, (short)0, c,
                                                false, false);
}

// A-matrix 16-bit 16x32 fragment: lane(0-15) row=lane, K = {kb..kb+15}
// with kb = 0 (lanes 0-15) or 16 (lanes 16-31).  p points at element K=kb of the row.
static __device__ __forceinline__ v16h load_a16(const _Float16* p) {
  v8h lo = *(const v8h*)p;
  v8h hh = *(const v8h*)(p + 8);
  v16h r;
#pragma unroll
  for (int j = 0; j < 8; ++j) { r[j] = lo[j]; r[j + 8] = hh[j]; }
  return r;
}

// ---- Kernel 0: weight f32->f16 + pre-pack into WMMA B-fragment order ----
// Packed layout: [kc][ntile][lane][i] ; element = W[kc*32 + (lane>>4)*16 + i]
//                                               [ntile*16 + (lane&15)]
// -> each lane's B fragment is one contiguous 32-byte run.
__global__ void wpack_kernel(const float* __restrict__ wq, const float* __restrict__ wl,
                             _Float16* __restrict__ wqp, _Float16* __restrict__ wlp) {
  int t = blockIdx.x * 256 + threadIdx.x;
  if (t < 6 * 36 * 32 * 16) {           // qkv weights: 6 kchunks x 36 ntiles
    int i = t & 15;
    int lane = (t >> 4) & 31;
    int nt = (t >> 9) % 36;
    int kc = (t >> 9) / 36;
    int k = kc * 32 + (lane >> 4) * 16 + i;
    int n = nt * 16 + (lane & 15);
    wqp[t] = (_Float16)wq[k * NQKV + n];
  }
  if (t < 6 * 12 * 32 * 16) {           // proj weights: 6 kchunks x 12 ntiles
    int i = t & 15;
    int lane = (t >> 4) & 31;
    int nt = (t >> 9) % 12;
    int kc = (t >> 9) / 12;
    int k = kc * 32 + (lane >> 4) * 16 + i;
    int n = nt * 16 + (lane & 15);
    wlp[t] = (_Float16)wl[k * CDIM + n];
  }
}

// ---- Kernel 1: fused cyclic-shift + QKV GEMM ----------------------------
// One block per window: 64 rows x 576 cols, K = 192. 12 waves, each wave
// computes 4 M-tiles x 3 N-tiles. Writes Q(prescaled)/K f16 row-major
// [wh][64][32] and V transposed [wh][32][64].
__global__ __launch_bounds__(384) void qkv_kernel(
    const float* __restrict__ x, const float* __restrict__ b_qkv,
    const _Float16* __restrict__ wqp,
    _Float16* __restrict__ Qb, _Float16* __restrict__ Kb, _Float16* __restrict__ Vb) {
  __shared__ __align__(16) _Float16 sX[64 * CDIM];   // 24576 B

  const int tid = threadIdx.x;
  const int blk = blockIdx.x;
  const int b   = blk >> 10;
  const int win = blk & 1023;
  const int w1  = win >> 5, w2 = win & 31;

  // Stage shifted x window (roll by -SHIFT => read at +4 mod 256), f32 -> f16.
#pragma unroll 1
  for (int i = tid; i < 64 * CDIM / 4; i += 384) {
    int e = i * 4;
    int row = e / CDIM, col = e % CDIM;
    int y  = (w1 * 8 + (row >> 3) + 4) & 255;
    int xx = (w2 * 8 + (row & 7) + 4) & 255;
    float4 v = *(const float4*)(x + (((size_t)b * 256 + y) * 256 + xx) * CDIM + col);
    _Float16* d = &sX[row * CDIM + col];
    d[0] = (_Float16)v.x; d[1] = (_Float16)v.y;
    d[2] = (_Float16)v.z; d[3] = (_Float16)v.w;
  }
  __syncthreads();

  const int wave = tid >> 5, lane = tid & 31;
  const int rl = lane & 15, hi = lane >> 4;

  v8f acc[4][3] = {};
#pragma unroll 1
  for (int kc = 0; kc < 6; ++kc) {
    v16h a[4];
#pragma unroll
    for (int mt = 0; mt < 4; ++mt)
      a[mt] = load_a16(&sX[(mt * 16 + rl) * CDIM + kc * 32 + hi * 16]);
#pragma unroll
    for (int nt = 0; nt < 3; ++nt) {
      // pre-packed B fragment: one contiguous 32B load per lane (L2-resident)
      v16h bf = *(const v16h*)(wqp + (((kc * 36 + (wave * 3 + nt)) * 32 + lane) << 4));
#pragma unroll
      for (int mt = 0; mt < 4; ++mt)
        acc[mt][nt] = wmma16(a[mt], bf, acc[mt][nt]);
    }
  }

  // Epilogue: bias, scale Q, scatter to per-window-head f16 buffers.
  const size_t wh6 = ((size_t)b * NWIN + win) * NH;
  const int n0 = wave * 48;
#pragma unroll
  for (int nt = 0; nt < 3; ++nt) {
    int n = n0 + nt * 16 + rl;
    int s = n / CDIM;
    int rem = n - s * CDIM;
    int head = rem >> 5, hd = rem & 31;
    float bias = b_qkv[n];
    float sc = (s == 0) ? SCALE_ : 1.0f;
    // Q,K: row-major [p][hd]; V: transposed [hd][p] so P*V B-frags are contiguous
    _Float16* dst;
    int stride;
    if (s == 2) { dst = Vb + (wh6 + head) * (size_t)(PWIN * HD) + hd * PWIN; stride = 1; }
    else {
      dst = ((s == 0) ? Qb : Kb) + (wh6 + head) * (size_t)(PWIN * HD) + hd;
      stride = HD;
    }
#pragma unroll
    for (int mt = 0; mt < 4; ++mt)
#pragma unroll
      for (int r = 0; r < 8; ++r) {
        int p = mt * 16 + hi * 8 + r;
        dst[p * stride] = (_Float16)((acc[mt][nt][r] + bias) * sc);
      }
  }
}

// ---- Kernel 2: windowed attention (one wave per window-head) ------------
__global__ __launch_bounds__(128) void attn_kernel(
    const _Float16* __restrict__ Qb, const _Float16* __restrict__ Kb,
    const _Float16* __restrict__ Vb, const float* __restrict__ rel_pos,
    _Float16* __restrict__ O) {
  __shared__ float sBias[225 * NH];                     // 5400 B
  __shared__ __align__(16) _Float16 sP[4][PWIN * PWIN]; // 32768 B

  const int tid = threadIdx.x;
#pragma unroll 1
  for (int i = tid; i < 225 * NH; i += 128) sBias[i] = rel_pos[i];
  __syncthreads();

  const int wave = tid >> 5, lane = tid & 31;
  const int rl = lane & 15, hi = lane >> 4;
  const int wh = blockIdx.x * 4 + wave;
  const int head = wh % NH;
  const int rest = wh / NH;
  const int win = rest & 1023;
  const int bidx = rest >> 10;
  const int w1 = win >> 5, w2 = win & 31;

  const _Float16* q = Qb + (size_t)wh * (PWIN * HD);
  const _Float16* k = Kb + (size_t)wh * (PWIN * HD);
  const _Float16* v = Vb + (size_t)wh * (PWIN * HD);   // transposed [32][64]

  // S = Q * K^T   (K-dim = 32 = head_dim, one WMMA per 16x16 tile)
  v16h qa[4];
#pragma unroll
  for (int mt = 0; mt < 4; ++mt)
    qa[mt] = load_a16(q + (mt * 16 + rl) * HD + hi * 16);

  v8f s[4][4] = {};
#pragma unroll
  for (int nt = 0; nt < 4; ++nt) {
    // B[c][q] = K[q][c] -> contiguous 16 halves per lane (32B aligned)
    v16h kb = *(const v16h*)(k + (nt * 16 + rl) * HD + hi * 16);
#pragma unroll
    for (int mt = 0; mt < 4; ++mt)
      s[mt][nt] = wmma16(qa[mt], kb, s[mt][nt]);
  }

  // bias + shifted-window mask + softmax (row lives in one 16-lane group)
  const bool mrow = (w1 == 31), mcol = (w2 == 31);
#pragma unroll
  for (int mt = 0; mt < 4; ++mt) {
#pragma unroll
    for (int r = 0; r < 8; ++r) {
      int p = mt * 16 + hi * 8 + r;
      int p1 = p >> 3, p2 = p & 7;
      float vals[4];
#pragma unroll
      for (int nt = 0; nt < 4; ++nt) {
        int qc = nt * 16 + rl;
        int q1 = qc >> 3, q2 = qc & 7;
        int idx = (p1 - q1 + 7) * 15 + (p2 - q2 + 7);
        bool masked = (mrow && ((p1 < 4) != (q1 < 4))) ||
                      (mcol && ((p2 < 4) != (q2 < 4)));
        vals[nt] = masked ? -1e30f : (s[mt][nt][r] + sBias[idx * NH + head]);
      }
      float m = fmaxf(fmaxf(vals[0], vals[1]), fmaxf(vals[2], vals[3]));
#pragma unroll
      for (int off = 1; off < 16; off <<= 1) m = fmaxf(m, __shfl_xor(m, off));
      float e[4], sum = 0.f;
#pragma unroll
      for (int nt = 0; nt < 4; ++nt) { e[nt] = __expf(vals[nt] - m); sum += e[nt]; }
#pragma unroll
      for (int off = 1; off < 16; off <<= 1) sum += __shfl_xor(sum, off);
      float inv = 1.0f / sum;
#pragma unroll
      for (int nt = 0; nt < 4; ++nt)
        sP[wave][p * PWIN + nt * 16 + rl] = (_Float16)(e[nt] * inv);
    }
  }
  __syncthreads();

  // Out = P * V   (V transposed: B-frag is contiguous 32B per lane)
  v8f o[4][2] = {};
#pragma unroll
  for (int kc = 0; kc < 2; ++kc) {
    v16h pa[4];
#pragma unroll
    for (int mt = 0; mt < 4; ++mt)
      pa[mt] = load_a16(&sP[wave][(mt * 16 + rl) * PWIN + kc * 32 + hi * 16]);
#pragma unroll
    for (int nt = 0; nt < 2; ++nt) {
      v16h vb = *(const v16h*)(v + (nt * 16 + rl) * PWIN + kc * 32 + hi * 16);
#pragma unroll
      for (int mt = 0; mt < 4; ++mt)
        o[mt][nt] = wmma16(pa[mt], vb, o[mt][nt]);
    }
  }

  // scatter f16 output rows back to shifted image layout [b][ys][xs][head*32+hd]
#pragma unroll
  for (int mt = 0; mt < 4; ++mt)
#pragma unroll
    for (int nt = 0; nt < 2; ++nt) {
      int hd = nt * 16 + rl;
#pragma unroll
      for (int r = 0; r < 8; ++r) {
        int p = mt * 16 + hi * 8 + r;
        int ys = w1 * 8 + (p >> 3);
        int xs = w2 * 8 + (p & 7);
        O[(((size_t)bidx * 256 + ys) * 256 + xs) * CDIM + head * HD + hd] =
            (_Float16)o[mt][nt][r];
      }
    }
}

// ---- Kernel 3: output projection + bias + un-shift roll -----------------
// One block = 64 consecutive rows of the shifted image, 12 waves each doing
// 4 M-tiles x 1 N-tile (16 cols), K = 192.  No LDS: B fragments come
// pre-packed from global (L2-resident), A fragments are contiguous rows.
__global__ __launch_bounds__(384) void proj_kernel(
    const _Float16* __restrict__ O, const _Float16* __restrict__ wlp,
    const float* __restrict__ b_lin, float* __restrict__ out) {
  const int tid = threadIdx.x, wave = tid >> 5, lane = tid & 31;
  const int rl = lane & 15, hi = lane >> 4;
  const size_t r0 = (size_t)blockIdx.x * 64;
  const int bidx = blockIdx.x >> 10;            // 1024 blocks per image
  const int rem = blockIdx.x & 1023;
  const int ys = rem >> 2;
  const int xs0 = (rem & 3) * 64;

  v8f acc[4] = {};
#pragma unroll 1
  for (int kc = 0; kc < 6; ++kc) {
    v16h bf = *(const v16h*)(wlp + (((kc * 12 + wave) * 32 + lane) << 4));
#pragma unroll
    for (int mt = 0; mt < 4; ++mt) {
      v16h a = load_a16(O + (r0 + mt * 16 + rl) * CDIM + kc * 32 + hi * 16);
      acc[mt] = wmma16(a, bf, acc[mt]);
    }
  }

  const int c = wave * 16 + rl;
  const float bias = b_lin[c];
#pragma unroll
  for (int mt = 0; mt < 4; ++mt)
#pragma unroll
    for (int r = 0; r < 8; ++r) {
      int xs = xs0 + mt * 16 + hi * 8 + r;
      int y  = (ys + 4) & 255;      // roll(+SHIFT)
      int xw = (xs + 4) & 255;
      out[(((size_t)bidx * 256 + y) * 256 + xw) * CDIM + c] = acc[mt][r] + bias;
    }
}

// ---- Host launcher -------------------------------------------------------
extern "C" void kernel_launch(void* const* d_in, const int* in_sizes, int n_in,
                              void* d_out, int out_size, void* d_ws, size_t ws_size,
                              hipStream_t stream) {
  const float* x       = (const float*)d_in[0];
  const float* w_qkv   = (const float*)d_in[1];
  const float* b_qkv   = (const float*)d_in[2];
  const float* w_lin   = (const float*)d_in[3];
  const float* b_lin   = (const float*)d_in[4];
  const float* rel_pos = (const float*)d_in[5];
  float* out = (float*)d_out;

  // workspace layout (f16 intermediates)
  char* ws = (char*)d_ws;
  const size_t SZ_WQ = (size_t)CDIM * NQKV * 2;               // 221184
  const size_t SZ_WL = (size_t)CDIM * CDIM * 2;               // 73728
  const size_t SZ_T  = (size_t)4 * NWIN * NH * PWIN * HD * 2; // 100663296 each
  _Float16* wqp = (_Float16*)ws;
  _Float16* wlp = (_Float16*)(ws + SZ_WQ);
  _Float16* Qb  = (_Float16*)(ws + SZ_WQ + SZ_WL);
  _Float16* Kb  = (_Float16*)(ws + SZ_WQ + SZ_WL + SZ_T);
  _Float16* Vb  = (_Float16*)(ws + SZ_WQ + SZ_WL + 2 * SZ_T);
  _Float16* Ob  = (_Float16*)(ws + SZ_WQ + SZ_WL + 3 * SZ_T);

  wpack_kernel<<<(CDIM * NQKV + 255) / 256, 256, 0, stream>>>(w_qkv, w_lin, wqp, wlp);
  qkv_kernel<<<4 * NWIN, 384, 0, stream>>>(x, b_qkv, wqp, Qb, Kb, Vb);
  attn_kernel<<<4 * NWIN * NH / 4, 128, 0, stream>>>(Qb, Kb, Vb, rel_pos, Ob);
  proj_kernel<<<4 * NWIN, 384, 0, stream>>>(Ob, wlp, b_lin, out);
}
